// NodeModel_42056319762634
// MI455X (gfx1250) — matrix operand order repaired and checked by reading, not compiled, not run on hardware
//
#include <hip/hip_runtime.h>
#include <hip/hip_bf16.h>

typedef __attribute__((ext_vector_type(16))) __bf16 v16bf;
typedef __attribute__((ext_vector_type(8)))  float  v8f;

#define NNODES_FEAT 64
#define EOUT 48
#define HID 128
#define KPAD 224   // 209 padded up to 7*32
#define MLPIN 209

struct __align__(16) U4 { unsigned int a, b, c, d; };
union FragA { U4 q[2]; v16bf v; };

static __device__ __forceinline__ unsigned short f2bf(float f) {
    unsigned u = __float_as_uint(f);
    unsigned r = ((u >> 16) & 1u) + 0x7fffu;   // round-to-nearest-even
    return (unsigned short)((u + r) >> 16);
}

// ---------------- phase 0: init scratch ----------------
__global__ void k_init(float* s, unsigned* mEnc, int* cnt, int n_nodes) {
    long long t = (long long)blockIdx.x * blockDim.x + threadIdx.x;
    long long tot = (long long)n_nodes * EOUT;
    if (t < tot) { s[t] = 0.0f; mEnc[t] = 0u; }
    if (t < n_nodes) cnt[t] = 0;
}

// ---------------- phase 1: edge scatter (sum / max / count) ----------------
// edge_attr / edge_index are streamed exactly once -> non-temporal loads so the
// 192MB L2 stays dedicated to the atomic targets (s/mEnc/cnt, ~38MB).
__global__ void k_scatter(const long long* __restrict__ ei, const float* __restrict__ ea,
                          float* s, unsigned* mEnc, int* cnt, long long n_edges) {
    long long t = (long long)blockIdx.x * blockDim.x + threadIdx.x;
    long long tot = n_edges * EOUT;
    if (t >= tot) return;
    long long e = t / EOUT;
    int j = (int)(t - e * EOUT);
    long long col = __builtin_nontemporal_load(&ei[n_edges + e]);   // destination node
    float v = __builtin_nontemporal_load(&ea[t]);                    // coalesced stream
    atomicAdd(&s[col * EOUT + j], v);
    unsigned u = __float_as_uint(v);
    unsigned key = (u & 0x80000000u) ? ~u : (u | 0x80000000u);       // monotonic encode
    atomicMax(&mEnc[col * EOUT + j], key);
    if (j == 0) atomicAdd(&cnt[col], 1);
}

// ---------------- phase 2a: weights -> bf16, transposed [N][K] ----------------
__global__ void k_convw(const float* __restrict__ W1, const float* __restrict__ W2,
                        unsigned short* w1t, unsigned short* w2t) {
    int t = blockIdx.x * blockDim.x + threadIdx.x;
    if (t < HID * KPAD) {                       // w1t[n][k], n<128, k<224
        int n = t / KPAD, k = t % KPAD;
        w1t[t] = f2bf(k < MLPIN ? W1[k * HID + n] : 0.0f);
    } else {
        int t2 = t - HID * KPAD;
        if (t2 < NNODES_FEAT * HID) {           // w2t[n][k], n<64, k<128
            int n = t2 / HID, k = t2 % HID;
            w2t[t2] = f2bf(W2[k * NNODES_FEAT + n]);
        }
    }
}

// ---------------- phase 2b: build concatenated h rows in bf16, K padded to 224 ----------------
__global__ void k_buildh(const float* __restrict__ x, const float* __restrict__ s,
                         const unsigned* __restrict__ mEnc, const int* __restrict__ cnt,
                         const float* __restrict__ u, const long long* __restrict__ batch,
                         unsigned short* hbf, int n_nodes) {
    long long t = (long long)blockIdx.x * blockDim.x + threadIdx.x;
    long long tot = (long long)n_nodes * KPAD;
    if (t >= tot) return;
    int node = (int)(t / KPAD);
    int k = (int)(t - (long long)node * KPAD);
    float v;
    if (k < 64) {
        v = x[(long long)node * 64 + k];
    } else if (k < 112) {
        v = s[(long long)node * EOUT + (k - 64)];
    } else if (k < 160) {
        if (cnt[node] > 0) {
            unsigned key = mEnc[(long long)node * EOUT + (k - 112)];
            v = (key & 0x80000000u) ? __uint_as_float(key ^ 0x80000000u)
                                    : __uint_as_float(~key);
        } else v = 0.0f;
    } else if (k < 208) {
        float c = (float)cnt[node];
        v = s[(long long)node * EOUT + (k - 160)] / fmaxf(c, 1.0f);
    } else if (k == 208) {
        v = u[batch[node]];
    } else {
        v = 0.0f;                               // pad 209..223
    }
    hbf[t] = f2bf(v);                           // stays L2-resident for the GEMM
}

// ---------------- phase 3: fused 2-layer MLP via v_wmma_f32_16x16x32_bf16 ----------------
// block = 256 threads = 8 waves; block handles 128 nodes, wave handles 16 nodes.
__global__ __launch_bounds__(256) void k_gemm(
    const unsigned short* __restrict__ hbf,   // [N][224] bf16
    const unsigned short* __restrict__ w1t,   // [128][224] bf16 (W1^T padded)
    const unsigned short* __restrict__ w2t,   // [64][128]  bf16 (W2^T)
    const float* __restrict__ b1, const float* __restrict__ b2,
    const float* __restrict__ x, float* __restrict__ out, int n_nodes)
{
    __shared__ __align__(16) unsigned short act[8][16][HID];   // 32 KB: relu(layer1) per wave

    const int lane = threadIdx.x & 31;
    const int wave = threadIdx.x >> 5;
    const int r  = lane & 15;      // A row (M) and B column (N)
    const int hi = lane >> 4;
    const int nodeBase = blockIdx.x * 128 + wave * 16;
    const int node = nodeBase + r;
    const long long nodeC = (node < n_nodes) ? node : 0;       // clamp for loads

    // ---- layer 1: [16x224] x [224x128] ----
    for (int nt = 0; nt < 8; ++nt) {
        v8f c = {};
        const unsigned short* brow0 = w1t + ((nt * 16 + r) * KPAD) + 16 * hi;
        const unsigned short* arow0 = hbf + nodeC * KPAD;
        #pragma unroll
        for (int kk = 0; kk < 7; ++kk) {
            FragA A, B;
            const unsigned short* arow = arow0 + kk * 32;
            A.q[0] = *(const U4*)(arow + 16 * hi);         // K [16*hi, +8)
            A.q[1] = *(const U4*)(arow + 16 * hi + 8);     // K [16*hi+8, +8)
            const unsigned short* brow = brow0 + kk * 32;
            B.q[0] = *(const U4*)(brow);                   // K [16*hi, +8)
            B.q[1] = *(const U4*)(brow + 8);               // K [16*hi+8, +8)
            c = __builtin_amdgcn_wmma_f32_16x16x32_bf16(false, A.v, false, B.v,
                                                        (short)0, c, false, false);
        }
        float bias = b1[nt * 16 + r];
        #pragma unroll
        for (int j = 0; j < 8; ++j) {
            float v = fmaxf(c[j] + bias, 0.0f);            // relu -> v_max_num_f32
            act[wave][j + 8 * hi][nt * 16 + r] = f2bf(v);  // C: row=j+8*hi, col=lane&15
        }
    }
    __syncthreads();

    // ---- layer 2: [16x128] x [128x64] + b2 + residual ----
    for (int nt = 0; nt < 4; ++nt) {
        v8f c = {};
        const unsigned short* brow0 = w2t + ((nt * 16 + r) * HID) + 16 * hi;
        #pragma unroll
        for (int kk = 0; kk < 4; ++kk) {
            FragA A, B;
            const unsigned short* arow = &act[wave][r][kk * 32];
            A.q[0] = *(const U4*)(arow + 16 * hi);
            A.q[1] = *(const U4*)(arow + 16 * hi + 8);
            const unsigned short* brow = brow0 + kk * 32;
            B.q[0] = *(const U4*)(brow);
            B.q[1] = *(const U4*)(brow + 8);
            c = __builtin_amdgcn_wmma_f32_16x16x32_bf16(false, A.v, false, B.v,
                                                        (short)0, c, false, false);
        }
        float bias = b2[nt * 16 + r];
        #pragma unroll
        for (int j = 0; j < 8; ++j) {
            int onode = nodeBase + j + 8 * hi;
            if (onode < n_nodes) {
                int cn = nt * 16 + r;
                long long idx = (long long)onode * NNODES_FEAT + cn;
                float res = c[j] + bias + x[idx];          // residual
                __builtin_nontemporal_store(res, &out[idx]);
            }
        }
    }
}

extern "C" void kernel_launch(void* const* d_in, const int* in_sizes, int n_in,
                              void* d_out, int out_size, void* d_ws, size_t ws_size,
                              hipStream_t stream) {
    const float*      x     = (const float*)d_in[0];
    const long long*  ei    = (const long long*)d_in[1];
    const float*      ea    = (const float*)d_in[2];
    const float*      u     = (const float*)d_in[3];
    const long long*  batch = (const long long*)d_in[4];
    const float*      W1    = (const float*)d_in[5];
    const float*      b1    = (const float*)d_in[6];
    const float*      W2    = (const float*)d_in[7];
    const float*      b2    = (const float*)d_in[8];

    const int       n_nodes = in_sizes[0] / NNODES_FEAT;
    const long long n_edges = (long long)in_sizes[1] / 2;

    auto al = [](size_t v) { return (v + 255) & ~(size_t)255; };
    char* base = (char*)d_ws;
    size_t off = 0;
    float*          s    = (float*)(base + off);          off += al((size_t)n_nodes * EOUT * 4);
    unsigned*       mEnc = (unsigned*)(base + off);       off += al((size_t)n_nodes * EOUT * 4);
    int*            cnt  = (int*)(base + off);            off += al((size_t)n_nodes * 4);
    unsigned short* w1t  = (unsigned short*)(base + off); off += al((size_t)HID * KPAD * 2);
    unsigned short* w2t  = (unsigned short*)(base + off); off += al((size_t)NNODES_FEAT * HID * 2);
    unsigned short* hbf  = (unsigned short*)(base + off); off += al((size_t)n_nodes * KPAD * 2);
    (void)ws_size;

    float* outp = (float*)d_out;

    {   // init sums / max-keys / counts
        long long tot = (long long)n_nodes * EOUT;
        k_init<<<(unsigned)((tot + 255) / 256), 256, 0, stream>>>(s, mEnc, cnt, n_nodes);
    }
    {   // scatter
        long long tot = n_edges * EOUT;
        k_scatter<<<(unsigned)((tot + 255) / 256), 256, 0, stream>>>(ei, ea, s, mEnc, cnt, n_edges);
    }
    {   // weights to bf16 transposed
        int tot = HID * KPAD + NNODES_FEAT * HID;
        k_convw<<<(tot + 255) / 256, 256, 0, stream>>>(W1, W2, w1t, w2t);
    }
    {   // build h rows (bf16, K padded to 224)
        long long tot = (long long)n_nodes * KPAD;
        k_buildh<<<(unsigned)((tot + 255) / 256), 256, 0, stream>>>(x, s, mEnc, cnt, u, batch, hbf, n_nodes);
    }
    {   // fused MLP with WMMA
        int blocks = (n_nodes + 127) / 128;
        k_gemm<<<blocks, 256, 0, stream>>>(hbf, w1t, w2t, b1, b2, x, outp, n_nodes);
    }
}
